// GenPhiloText_14379550507654
// MI455X (gfx1250) — matrix operand order repaired and checked by reading, not compiled, not run on hardware
//
#include <hip/hip_runtime.h>
#include <hip/hip_bf16.h>

// Problem dims (fixed by the reference)
#define B_  1024
#define T_  50
#define V_  256
#define H_  1024

typedef __bf16 bf16;
typedef __attribute__((ext_vector_type(16))) __bf16        v16bf;
typedef __attribute__((ext_vector_type(8)))  float         v8f;
typedef __attribute__((ext_vector_type(4)))  unsigned int  v4u;

union FragBF {
    struct { v4u lo; v4u hi; } u;
    v16bf v;
};

// ---------------------------------------------------------------------------
// A-matrix fragment (16x32 bf16, MxK), contiguous K order (not the ISA table order):
//   lanes 0-15 : row M=lane, K = 0..15
//   lanes 16-31: row M=lane-16, K = 16..31
// Source is row-major [rows][K] bf16, viewed as dwords (2 bf16 per dword).
// Per lane this is two contiguous 16-byte chunks -> 2x global_load_b128.
// ---------------------------------------------------------------------------
__device__ __forceinline__ v16bf load_fragA(const unsigned int* __restrict__ base_dw,
                                            int ld_dw, int row0, int k0_dw, int lane) {
    const int half = lane >> 4;
    const int r    = lane & 15;
    const unsigned int* p = base_dw + (size_t)(row0 + r) * ld_dw + k0_dw + 8 * half;
    FragBF f;
    f.u.lo = *(const v4u*)(p);       // K = k0 + 16*half + 0..7
    f.u.hi = *(const v4u*)(p + 4);   // K = k0 + 16*half + 8..15
    return f.v;
}

// ---------------------------------------------------------------------------
// B-matrix fragment (32x16 bf16, KxN), per the K-striped B layout:
//   lanes 0-15 : col N=lane, K = 0..15 (VGPR j -> K pair {2j,2j+1})
//   lanes 16-31: col N=lane-16, K = 16..31
// Source is the *transposed* weight, row-major [N][K] bf16 -> per lane one
// contiguous 32-byte run -> 2x global_load_b128.
// ---------------------------------------------------------------------------
__device__ __forceinline__ v16bf load_fragB(const unsigned int* __restrict__ base_dw,
                                            int ld_dw, int n0, int k0_dw, int lane) {
    const int half = lane >> 4;
    const int n    = lane & 15;
    const unsigned int* p = base_dw + (size_t)(n0 + n) * ld_dw + k0_dw + 8 * half;
    FragBF f;
    f.u.lo = *(const v4u*)(p);       // K = k0 + 16*half + 0..7
    f.u.hi = *(const v4u*)(p + 4);   // K = k0 + 16*half + 8..15
    return f.v;
}

#define WMMA_BF16(A, Bm, C) \
    __builtin_amdgcn_wmma_f32_16x16x32_bf16(false, (A), false, (Bm), (short)0, (C), false, false)

__device__ __forceinline__ float sigmoidf_(float x) {
    return 1.0f / (1.0f + __expf(-x));
}

// ---------------------------------------------------------------------------
// Conversion / init kernels (run once per launch; deterministic, stateless)
// ---------------------------------------------------------------------------
__global__ void cvt_bf16_kernel(const float* __restrict__ in, bf16* __restrict__ out, int n) {
    int i = blockIdx.x * blockDim.x + threadIdx.x;
    if (i < n) out[i] = (bf16)in[i];
}

// out (row-major [rows_out][cols_out] bf16) = transpose of in ([cols_out][rows_out] f32)
__global__ void transpose_cvt_kernel(const float* __restrict__ in, bf16* __restrict__ out,
                                     int rows_out, int cols_out) {
    int i = blockIdx.x * blockDim.x + threadIdx.x;
    int total = rows_out * cols_out;
    if (i < total) {
        int n = i / cols_out;
        int k = i - n * cols_out;
        out[i] = (bf16)in[(size_t)k * rows_out + n];
    }
}

__global__ void init_state_kernel(const float* __restrict__ h0, const float* __restrict__ c0,
                                  bf16* __restrict__ hbuf, float* __restrict__ c, int n) {
    int i = blockIdx.x * blockDim.x + threadIdx.x;
    if (i < n) { hbuf[i] = (bf16)h0[i]; c[i] = c0[i]; }
}

// ---------------------------------------------------------------------------
// Fused LSTM step: z = x_t @ W_x + h @ W_h + b for all 4 gates, then the cell
// update, entirely in registers per lane.
// Grid: (B/64, H/64), block 256 (8 waves). Wave (wmg=w>>2, wng=w&3) owns a
// 32x16 tile at (m0 = bx*64 + wmg*32, n0 = by*64 + wng*16) for each gate.
// ---------------------------------------------------------------------------
__global__ __launch_bounds__(256) void lstm_step_kernel(
    const unsigned int* __restrict__ Xbf,   // [B][T*V] bf16 (dword view)
    const unsigned int* __restrict__ Wxt,   // [4H][V]  bf16 (dword view), transposed W_x
    const unsigned int* __restrict__ Wht,   // [4H][H]  bf16 (dword view), transposed W_h
    const unsigned int* __restrict__ hin,   // [B][H]   bf16 (dword view)
    bf16* __restrict__ hout,                // [B][H]   bf16
    float* __restrict__ c,                  // [B][H]   f32 (in/out, unique owner per elem)
    const float* __restrict__ bias,         // [4H]
    int t)
{
    const int lane = threadIdx.x & 31;
    const int w    = threadIdx.x >> 5;
    const int wmg  = w >> 2;             // 0..1
    const int wng  = w & 3;              // 0..3
    const int m0   = blockIdx.x * 64 + wmg * 32;
    const int n0   = blockIdx.y * 64 + wng * 16;

    v8f acc[4][2];
    #pragma unroll
    for (int g = 0; g < 4; ++g) { acc[g][0] = (v8f)0.0f; acc[g][1] = (v8f)0.0f; }

    // ---- Phase 1: x_t @ W_x  (K = V = 256) ----
    const unsigned int* Xrow = Xbf + (size_t)t * (V_ / 2);  // dword offset of time-slice
    const int ldX = (T_ * V_) / 2;
    #pragma unroll 1
    for (int k0 = 0; k0 < V_ / 2; k0 += 16) {               // k0 in dwords (32 K elems)
        v16bf a0 = load_fragA(Xrow, ldX, m0,      k0, lane);
        v16bf a1 = load_fragA(Xrow, ldX, m0 + 16, k0, lane);
        #pragma unroll
        for (int g = 0; g < 4; ++g) {
            v16bf bb = load_fragB(Wxt, V_ / 2, g * H_ + n0, k0, lane);
            acc[g][0] = WMMA_BF16(a0, bb, acc[g][0]);
            acc[g][1] = WMMA_BF16(a1, bb, acc[g][1]);
        }
    }

    // ---- Phase 2: h @ W_h  (K = H = 1024) ----
    #pragma unroll 1
    for (int k0 = 0; k0 < H_ / 2; k0 += 16) {
        v16bf a0 = load_fragA(hin, H_ / 2, m0,      k0, lane);
        v16bf a1 = load_fragA(hin, H_ / 2, m0 + 16, k0, lane);
        #pragma unroll
        for (int g = 0; g < 4; ++g) {
            v16bf bb = load_fragB(Wht, H_ / 2, g * H_ + n0, k0, lane);
            acc[g][0] = WMMA_BF16(a0, bb, acc[g][0]);
            acc[g][1] = WMMA_BF16(a1, bb, acc[g][1]);
        }
    }

    // ---- Gate nonlinearities + cell update (i,f,g,o live in this lane) ----
    const int half = lane >> 4;
    const int nr   = lane & 15;
    const int n    = n0 + nr;
    const float bi = bias[n];
    const float bf_v = bias[H_ + n];
    const float bg = bias[2 * H_ + n];
    const float bo = bias[3 * H_ + n];

    #pragma unroll
    for (int sm = 0; sm < 2; ++sm) {
        #pragma unroll
        for (int r = 0; r < 8; ++r) {
            const int m = m0 + sm * 16 + r + 8 * half;       // C/D layout: M = r + 8*half
            const size_t idx = (size_t)m * H_ + n;
            const float ig = sigmoidf_(acc[0][sm][r] + bi);
            const float fg = sigmoidf_(acc[1][sm][r] + bf_v);
            const float gg = tanhf    (acc[2][sm][r] + bg);
            const float og = sigmoidf_(acc[3][sm][r] + bo);
            const float cn = fg * c[idx] + ig * gg;
            const float hn = og * tanhf(cn);
            c[idx]    = cn;
            hout[idx] = (bf16)hn;
        }
    }
}

// ---------------------------------------------------------------------------
// Readout GEMM: logits = h_new @ W_d + b_d   ([B,H]x[H,V] -> [B,V])
// Grid: (B/64, V/64), block 256 (8 waves), same 32x16 wave tile.
// ---------------------------------------------------------------------------
__global__ __launch_bounds__(256) void readout_kernel(
    const unsigned int* __restrict__ hbf,   // [B][H] bf16 (dword view)
    const unsigned int* __restrict__ Wdt,   // [V][H] bf16 (dword view), transposed W_d
    const float* __restrict__ bd,           // [V]
    float* __restrict__ logits)             // [B][V] f32
{
    const int lane = threadIdx.x & 31;
    const int w    = threadIdx.x >> 5;
    const int wmg  = w >> 2;
    const int wng  = w & 3;
    const int m0   = blockIdx.x * 64 + wmg * 32;
    const int n0   = blockIdx.y * 64 + wng * 16;

    v8f acc0 = (v8f)0.0f, acc1 = (v8f)0.0f;
    #pragma unroll 1
    for (int k0 = 0; k0 < H_ / 2; k0 += 16) {
        v16bf a0 = load_fragA(hbf, H_ / 2, m0,      k0, lane);
        v16bf a1 = load_fragA(hbf, H_ / 2, m0 + 16, k0, lane);
        v16bf bb = load_fragB(Wdt, H_ / 2, n0,      k0, lane);
        acc0 = WMMA_BF16(a0, bb, acc0);
        acc1 = WMMA_BF16(a1, bb, acc1);
    }

    const int half = lane >> 4;
    const int n    = n0 + (lane & 15);
    const float bn = bd[n];
    #pragma unroll
    for (int r = 0; r < 8; ++r) {
        int m = m0 + r + 8 * half;
        logits[(size_t)m * V_ + n] = acc0[r] + bn;
        m += 16;
        logits[(size_t)m * V_ + n] = acc1[r] + bn;
    }
}

// ---------------------------------------------------------------------------
// Row softmax over V=256 and scatter into out[b, t, :]. One block per row.
// ---------------------------------------------------------------------------
__global__ __launch_bounds__(256) void softmax_kernel(
    const float* __restrict__ logits, float* __restrict__ out, int t)
{
    __shared__ float red[256];
    const int b = blockIdx.x;
    const int v = threadIdx.x;
    const float x = logits[(size_t)b * V_ + v];
    red[v] = x; __syncthreads();
    for (int s = 128; s > 0; s >>= 1) {
        if (v < s) red[v] = fmaxf(red[v], red[v + s]);
        __syncthreads();
    }
    const float mx = red[0]; __syncthreads();
    const float e = __expf(x - mx);
    red[v] = e; __syncthreads();
    for (int s = 128; s > 0; s >>= 1) {
        if (v < s) red[v] += red[v + s];
        __syncthreads();
    }
    out[(size_t)b * T_ * V_ + (size_t)t * V_ + v] = e / red[0];
}

// ---------------------------------------------------------------------------
// Host-side orchestration. Inputs (setup_inputs order):
//   0:X [B,T,V]  1:h_0 [B,H]  2:c_0 [B,H]  3:W_x [V,4H]  4:W_h [H,4H]
//   5:b [4H]     6:W_d [H,V]  7:b_d [V]
// Workspace use: ~46.7 MB.
// ---------------------------------------------------------------------------
extern "C" void kernel_launch(void* const* d_in, const int* in_sizes, int n_in,
                              void* d_out, int out_size, void* d_ws, size_t ws_size,
                              hipStream_t stream) {
    (void)in_sizes; (void)n_in; (void)out_size; (void)ws_size;
    const float* X   = (const float*)d_in[0];
    const float* h0  = (const float*)d_in[1];
    const float* c0  = (const float*)d_in[2];
    const float* Wx  = (const float*)d_in[3];
    const float* Wh  = (const float*)d_in[4];
    const float* bb  = (const float*)d_in[5];
    const float* Wd  = (const float*)d_in[6];
    const float* bd  = (const float*)d_in[7];
    float* out = (float*)d_out;

    char* ws = (char*)d_ws;
    size_t off = 0;
    auto alloc = [&](size_t bytes) -> void* {
        void* p = ws + off;
        off += (bytes + 255) & ~(size_t)255;
        return p;
    };
    bf16*  Xbf   = (bf16*) alloc((size_t)B_ * T_ * V_ * 2);   // 26.2 MB
    bf16*  Wxt   = (bf16*) alloc((size_t)4 * H_ * V_ * 2);    //  2.1 MB
    bf16*  Wht   = (bf16*) alloc((size_t)4 * H_ * H_ * 2);    //  8.4 MB
    bf16*  Wdt   = (bf16*) alloc((size_t)V_ * H_ * 2);        //  0.5 MB
    bf16*  hbuf0 = (bf16*) alloc((size_t)B_ * H_ * 2);        //  2.1 MB
    bf16*  hbuf1 = (bf16*) alloc((size_t)B_ * H_ * 2);        //  2.1 MB
    float* cbuf  = (float*)alloc((size_t)B_ * H_ * 4);        //  4.2 MB
    float* logit = (float*)alloc((size_t)B_ * V_ * 4);        //  1.0 MB

    // One-time (per call) precision conversion + weight transposes.
    {
        const int nX = B_ * T_ * V_;
        cvt_bf16_kernel<<<(nX + 255) / 256, 256, 0, stream>>>(X, Xbf, nX);
        const int nWx = 4 * H_ * V_;
        transpose_cvt_kernel<<<(nWx + 255) / 256, 256, 0, stream>>>(Wx, Wxt, 4 * H_, V_);
        const int nWh = 4 * H_ * H_;
        transpose_cvt_kernel<<<(nWh + 255) / 256, 256, 0, stream>>>(Wh, Wht, 4 * H_, H_);
        const int nWd = V_ * H_;
        transpose_cvt_kernel<<<(nWd + 255) / 256, 256, 0, stream>>>(Wd, Wdt, V_, H_);
        const int nS = B_ * H_;
        init_state_kernel<<<(nS + 255) / 256, 256, 0, stream>>>(h0, c0, hbuf0, cbuf, nS);
    }

    // Sequential scan over time; h double-buffered across steps.
    const dim3 gGate(B_ / 64, H_ / 64);   // (16,16)
    const dim3 gRead(B_ / 64, V_ / 64);   // (16,4)
    for (int t = 0; t < T_; ++t) {
        bf16* hin  = (t & 1) ? hbuf1 : hbuf0;
        bf16* hout = (t & 1) ? hbuf0 : hbuf1;
        lstm_step_kernel<<<gGate, 256, 0, stream>>>(
            (const unsigned int*)Xbf, (const unsigned int*)Wxt,
            (const unsigned int*)Wht, (const unsigned int*)hin,
            hout, cbuf, bb, t);
        readout_kernel<<<gRead, 256, 0, stream>>>(
            (const unsigned int*)hout, (const unsigned int*)Wdt, bd, logit);
        softmax_kernel<<<B_, 256, 0, stream>>>(logit, out, t);
    }
}
